// Mamba3TextEncoder_88536455840132
// MI455X (gfx1250) — matrix operand order repaired and checked by reading, not compiled, not run on hardware
//
#include <hip/hip_runtime.h>
#include <hip/hip_bf16.h>
#include <math.h>

// ---------------- problem constants ----------------
#define B_SZ     4
#define SEQ      512
#define NTOK     (B_SZ * SEQ)          // 2048
#define D_MODEL  768
#define N_LAYERS 12
#define N_STATE  64
#define HEADDIM  32
#define NHEADS   48
#define D_INNER  1536
#define CONV_DIM 1664                  // D_INNER + 2*64
#define K_CONV   4
#define PROJ_IN  3248                  // 2*D_INNER + 2*64 + NHEADS
#define EMBED    1536
#define NCHUNK   16
#define CHUNK    32                    // SEQ / NCHUNK
#define STATE_SZ (HEADDIM * N_STATE)   // 2048 f32 per (b,head)

// ---------------- WMMA vector types ----------------
typedef __attribute__((ext_vector_type(16))) __bf16 v16bf;
typedef __attribute__((ext_vector_type(8)))  __bf16 v8bf;
typedef __attribute__((ext_vector_type(8)))  float  v8f;

// ---------------- helpers ----------------
__device__ __forceinline__ unsigned short to_bf16_bits(float f) {
    unsigned int u = __float_as_uint(f);
    u += 0x7FFFu + ((u >> 16) & 1u);   // round-to-nearest-even
    return (unsigned short)(u >> 16);
}

__device__ __forceinline__ float silu_f(float x) {
    return x / (1.0f + __expf(-x));
}

__device__ __forceinline__ float block_reduce_sum(float v, float* red) {
    int t = threadIdx.x;
    red[t] = v;
    __syncthreads();
    for (int s = blockDim.x >> 1; s > 0; s >>= 1) {
        if (t < s) red[t] += red[t + s];
        __syncthreads();
    }
    float r = red[0];
    __syncthreads();
    return r;
}

// CDNA5 async global->LDS copy (ASYNCcnt-tracked), one b128 per lane.
// dst operand = VGPR with LDS byte address (HW truncates flat addr to [31:0]).
__device__ __forceinline__ void async_copy_b128(const void* gaddr, void* lds_ptr) {
    unsigned lds_off = (unsigned)(size_t)lds_ptr;
    unsigned long long ga = (unsigned long long)(size_t)gaddr;
    asm volatile("global_load_async_to_lds_b128 %0, %1, off"
                 :: "v"(lds_off), "v"(ga) : "memory");
}
__device__ __forceinline__ void wait_async_le1() {
    asm volatile("s_wait_asynccnt 0x1" ::: "memory");
}
__device__ __forceinline__ void wait_async_le0() {
    asm volatile("s_wait_asynccnt 0x0" ::: "memory");
}
// Bare workgroup split-barrier: no implicit vmem-drain fence (unlike
// __syncthreads), so in-flight next-tile loads stay in flight across it.
__device__ __forceinline__ void wg_barrier() {
    asm volatile("s_barrier_signal -1\n\ts_barrier_wait -1" ::: "memory");
}

// Stage one 16x32 bf16 A tile (1KB) into LDS: threads 0..63, 16B each.
__device__ __forceinline__ void stage_A_async(const unsigned short* A, int ld,
                                              int mrow, int k0,
                                              unsigned short* lbuf) {
    int tid = threadIdx.x;
    if (tid < 64) {
        int row = tid >> 2, seg = tid & 3;                 // 4 x 8-elem segs per row
        const unsigned short* g = A + (size_t)(mrow + row) * ld + k0 + seg * 8;
        async_copy_b128(g, lbuf + row * 32 + seg * 8);
    }
}

// B fragment from transposed weight WT[N x K]: lane<16 -> col=lane, K contiguous
__device__ __forceinline__ v16bf load_fragB(const __bf16* browK) {
    return *(const v16bf*)browK;
}

// A fragment from an LDS 16x32 tile: lane<16 K={0..15}, else K={16..31}
__device__ __forceinline__ v16bf load_fragA_lds(const unsigned short* lbuf,
                                                int cl, int h) {
    const __bf16* ap = (const __bf16*)(lbuf + cl * 32 + h * 16);
    v8bf lo = *(const v8bf*)ap;
    v8bf hi = *(const v8bf*)(ap + 8);
    v16bf a;
#pragma unroll
    for (int i = 0; i < 8; ++i) { a[i] = lo[i]; a[8 + i] = hi[i]; }
    return a;
}

// ---------------- kernels ----------------

// x[t,d] = embed_W[ids[t], d]
__global__ void k_embed(const int* __restrict__ ids, const float* __restrict__ embW,
                        float* __restrict__ x) {
    int idx = blockIdx.x * blockDim.x + threadIdx.x;
    if (idx >= NTOK * D_MODEL) return;
    int t = idx / D_MODEL, d = idx - t * D_MODEL;
    x[idx] = embW[(size_t)ids[t] * D_MODEL + d];
}

// per-token RMSNorm over D, bf16 output (feeds WMMA GEMM). 1 block / token.
__global__ void __launch_bounds__(256)
k_rmsnorm_bf16(const float* __restrict__ x, const float* __restrict__ w,
               unsigned short* __restrict__ out, int D) {
    __shared__ float red[256];
    int t = blockIdx.x;
    const float* xr = x + (size_t)t * D;
    float ss = 0.f;
    for (int d = threadIdx.x; d < D; d += blockDim.x) { float v = xr[d]; ss += v * v; }
    ss = block_reduce_sum(ss, red);
    float scale = rsqrtf(ss / (float)D + 1e-5f);
    unsigned short* orow = out + (size_t)t * D;
    for (int d = threadIdx.x; d < D; d += blockDim.x)
        orow[d] = to_bf16_bits(xr[d] * scale * w[d]);
}

// per-token RMSNorm over D, f32 output (final norm)
__global__ void __launch_bounds__(256)
k_rmsnorm_f32(const float* __restrict__ x, const float* __restrict__ w,
              float* __restrict__ out, int D) {
    __shared__ float red[256];
    int t = blockIdx.x;
    const float* xr = x + (size_t)t * D;
    float ss = 0.f;
    for (int d = threadIdx.x; d < D; d += blockDim.x) { float v = xr[d]; ss += v * v; }
    ss = block_reduce_sum(ss, red);
    float scale = rsqrtf(ss / (float)D + 1e-5f);
    float* orow = out + (size_t)t * D;
    for (int d = threadIdx.x; d < D; d += blockDim.x)
        orow[d] = xr[d] * scale * w[d];
}

// LDS-tiled transpose+convert: W[K x N] f32 -> WT[N x K] bf16 (coalesced both sides)
__global__ void __launch_bounds__(256)
k_transpose_bf16(const float* __restrict__ W, unsigned short* __restrict__ WT,
                 int K, int N) {
    __shared__ unsigned short tile[32][33];
    int nb = blockIdx.x * 32, kb = blockIdx.y * 32;
    int tx = threadIdx.x & 31, ty = threadIdx.x >> 5;    // ty = 0..7
    for (int i = ty; i < 32; i += 8) {
        int k = kb + i, n = nb + tx;
        tile[i][tx] = (k < K && n < N) ? to_bf16_bits(W[(size_t)k * N + n]) : 0;
    }
    __syncthreads();
    for (int i = ty; i < 32; i += 8) {
        int n = nb + i, k = kb + tx;
        if (n < N && k < K) WT[(size_t)n * K + k] = tile[tx][i];
    }
}

// C[M x N](f32) = A[M x K](bf16 row-major) @ BT[N x K](bf16)^T (+ optional residual)
// 4 waves/block; wave tile = 16M x 64N (4 accumulators per A fragment).
// A tile double-buffered in LDS via async-to-LDS; B fragments ping-pong
// pipelined (K-loop unrolled x2, bare split barriers so in-flight loads
// are not drained at the barrier). Requires K % 64 == 0 (768, 1536 here).
__global__ void __launch_bounds__(128)
k_gemm_bf16(const unsigned short* __restrict__ A, const unsigned short* __restrict__ BT,
            const float* __restrict__ resid, float* __restrict__ C,
            int M, int N, int K) {
    __shared__ __align__(16) unsigned short Atile[2][16 * 32];
    (void)M;
    int lane  = threadIdx.x & 31;
    int wave  = threadIdx.x >> 5;
    int mrow  = blockIdx.y * 16;
    int ncol0 = blockIdx.x * 256 + wave * 64;

    int h = lane >> 4, cl = lane & 15;
    // per-subtile B row base pointers (clamped for OOB; stores guarded below)
    const __bf16* browp[4];
#pragma unroll
    for (int j = 0; j < 4; ++j) {
        int col = ncol0 + j * 16;
        if (col > N - 16) col = N - 16;
        browp[j] = (const __bf16*)BT + (size_t)(col + cl) * K + h * 16;
    }

    int kt = K >> 5;                       // even
    stage_A_async(A, K, mrow, 0, Atile[0]);

    v16bf bc0[4], bc1[4];
#pragma unroll
    for (int j = 0; j < 4; ++j) bc0[j] = load_fragB(browp[j]);

    v8f acc[4] = {{}, {}, {}, {}};

    for (int kk = 0; kk < kt; kk += 2) {
        // ---- step kk : compute from Atile[0] / bc0, prefetch kk+1 ----
        stage_A_async(A, K, mrow, (kk + 1) << 5, Atile[1]);
#pragma unroll
        for (int j = 0; j < 4; ++j) bc1[j] = load_fragB(browp[j] + ((kk + 1) << 5));
        wait_async_le1();                  // stage(kk) done (in-order), stage(kk+1) in flight
        wg_barrier();
        {
            v16bf a = load_fragA_lds(Atile[0], cl, h);
#pragma unroll
            for (int j = 0; j < 4; ++j)
                acc[j] = __builtin_amdgcn_wmma_f32_16x16x32_bf16(
                             false, a, false, bc0[j], (short)0, acc[j], false, false);
        }
        wg_barrier();                      // all reads of Atile[0] done before reuse

        // ---- step kk+1 : compute from Atile[1] / bc1, prefetch kk+2 ----
        if (kk + 2 < kt) {
            stage_A_async(A, K, mrow, (kk + 2) << 5, Atile[0]);
#pragma unroll
            for (int j = 0; j < 4; ++j) bc0[j] = load_fragB(browp[j] + ((kk + 2) << 5));
            wait_async_le1();
        } else {
            wait_async_le0();
        }
        wg_barrier();
        {
            v16bf a = load_fragA_lds(Atile[1], cl, h);
#pragma unroll
            for (int j = 0; j < 4; ++j)
                acc[j] = __builtin_amdgcn_wmma_f32_16x16x32_bf16(
                             false, a, false, bc1[j], (short)0, acc[j], false, false);
        }
        wg_barrier();
    }

    // D layout: VGPR i -> row = i + (lane>=16 ? 8 : 0), col = lane & 15
    int rbase = mrow + (h << 3);
#pragma unroll
    for (int j = 0; j < 4; ++j) {
        int col0 = ncol0 + j * 16;
        if (col0 >= N) continue;
        int col = col0 + cl;
#pragma unroll
        for (int i = 0; i < 8; ++i) {
            size_t off = (size_t)(rbase + i) * N + col;
            float v = acc[j][i];
            if (resid) v += resid[off];
            C[off] = v;
        }
    }
}

// depthwise causal conv (K=4) + bias + SiLU over xBC slice of zxbcdt
__global__ void k_conv_silu(const float* __restrict__ zx, const float* __restrict__ cw,
                            const float* __restrict__ cb, float* __restrict__ xbc) {
    int idx = blockIdx.x * blockDim.x + threadIdx.x;
    if (idx >= NTOK * CONV_DIM) return;
    int t = idx / CONV_DIM, c = idx - t * CONV_DIM;
    int b = t / SEQ, s = t - b * SEQ;
    float acc = cb[c];
#pragma unroll
    for (int k = 0; k < K_CONV; ++k) {
        int sp = s - (K_CONV - 1) + k;
        if (sp >= 0)
            acc += zx[(size_t)(b * SEQ + sp) * PROJ_IN + D_INNER + c] * cw[c * K_CONV + k];
    }
    xbc[idx] = silu_f(acc);
}

// dt = softplus(dt_raw + dt_bias)
__global__ void k_dt(const float* __restrict__ zx, const float* __restrict__ dtb,
                     float* __restrict__ dt) {
    int idx = blockIdx.x * blockDim.x + threadIdx.x;
    if (idx >= NTOK * NHEADS) return;
    int t = idx / NHEADS, hh = idx - t * NHEADS;
    float v = zx[(size_t)t * PROJ_IN + D_INNER + CONV_DIM + hh] + dtb[hh];
    dt[idx] = (v > 20.f) ? v : log1pf(__expf(v));
}

// ---- chunked SSM scan (exact: diagonal-linear recurrence) ----
// Phase A: per-chunk local scan from zero state. grid = B*NHEADS*NCHUNK.
__global__ void __launch_bounds__(256)
k_ssm_chunk(const float* __restrict__ xbc, const float* __restrict__ dt,
            const float* __restrict__ A_log, const float* __restrict__ Dvec,
            float* __restrict__ y, float* __restrict__ cumdec,
            float* __restrict__ Sbuf) {
    int blk = blockIdx.x;
    int c  = blk & (NCHUNK - 1);
    int bh = blk >> 4;                       // NCHUNK == 16
    int b = bh / NHEADS, hd = bh - b * NHEADS;
    float A  = -__expf(A_log[hd]);
    float Dv = Dvec[hd];
    int p  = threadIdx.x >> 3;
    int nc = threadIdx.x & 7;

    float h[8];
#pragma unroll
    for (int i = 0; i < 8; ++i) h[i] = 0.f;
    float run = 1.f;

    int s0 = c * CHUNK;
    for (int s = s0; s < s0 + CHUNK; ++s) {
        int tok = b * SEQ + s;
        const float* row = xbc + (size_t)tok * CONV_DIM;
        float dtv  = dt[(size_t)tok * NHEADS + hd];
        float xs_p = row[hd * HEADDIM + p];
        float dec  = __expf(dtv * A);
        float dx   = dtv * xs_p;
        run *= dec;
        if (threadIdx.x == 0) cumdec[(size_t)tok * NHEADS + hd] = run;
        const float4* Bp = (const float4*)(row + D_INNER + nc * 8);
        const float4* Cp = (const float4*)(row + D_INNER + N_STATE + nc * 8);
        float4 b0 = Bp[0], b1 = Bp[1];
        float4 c0 = Cp[0], c1 = Cp[1];
        float acc = 0.f;
        h[0] = dec * h[0] + dx * b0.x; acc += h[0] * c0.x;
        h[1] = dec * h[1] + dx * b0.y; acc += h[1] * c0.y;
        h[2] = dec * h[2] + dx * b0.z; acc += h[2] * c0.z;
        h[3] = dec * h[3] + dx * b0.w; acc += h[3] * c0.w;
        h[4] = dec * h[4] + dx * b1.x; acc += h[4] * c1.x;
        h[5] = dec * h[5] + dx * b1.y; acc += h[5] * c1.y;
        h[6] = dec * h[6] + dx * b1.z; acc += h[6] * c1.z;
        h[7] = dec * h[7] + dx * b1.w; acc += h[7] * c1.w;
        acc += __shfl_xor(acc, 1, 32);
        acc += __shfl_xor(acc, 2, 32);
        acc += __shfl_xor(acc, 4, 32);
        if (nc == 0)
            y[(size_t)tok * D_INNER + hd * HEADDIM + p] = acc + Dv * xs_p;
    }
    float* Sp = Sbuf + ((size_t)bh * NCHUNK + c) * STATE_SZ + p * N_STATE + nc * 8;
#pragma unroll
    for (int i = 0; i < 8; ++i) Sp[i] = h[i];
}

// Phase B: sequential combine over 16 chunks; Sbuf rewritten in place with
// H_init[c] (state entering chunk c). grid = B*NHEADS, 256 threads x 8 elems.
__global__ void __launch_bounds__(256)
k_ssm_combine(float* __restrict__ Sbuf, const float* __restrict__ cumdec) {
    int bh = blockIdx.x;
    int b = bh / NHEADS, hd = bh - b * NHEADS;
    float H[8];
#pragma unroll
    for (int i = 0; i < 8; ++i) H[i] = 0.f;
    float* base = Sbuf + (size_t)bh * NCHUNK * STATE_SZ + threadIdx.x * 8;
    for (int c = 0; c < NCHUNK; ++c) {
        float* Sp = base + (size_t)c * STATE_SZ;
        float P = cumdec[(size_t)(b * SEQ + c * CHUNK + CHUNK - 1) * NHEADS + hd];
        float s[8];
#pragma unroll
        for (int i = 0; i < 8; ++i) s[i] = Sp[i];
#pragma unroll
        for (int i = 0; i < 8; ++i) Sp[i] = H[i];      // store H_init[c]
#pragma unroll
        for (int i = 0; i < 8; ++i) H[i] = P * H[i] + s[i];
    }
}

// Phase C: y_t += cumdec_t * (C_t . H_init[chunk(t)]).  grid = NTOK*NHEADS.
__global__ void __launch_bounds__(256)
k_ssm_fix(const float* __restrict__ xbc, const float* __restrict__ cumdec,
          const float* __restrict__ Hinit, float* __restrict__ y) {
    int blk = blockIdx.x;
    int tok = blk / NHEADS, hd = blk - tok * NHEADS;
    int b = tok / SEQ, s = tok - b * SEQ;
    int c = s / CHUNK;
    int p = threadIdx.x >> 3, nc = threadIdx.x & 7;
    const float* Crow = xbc + (size_t)tok * CONV_DIM + D_INNER + N_STATE;
    const float* Hp = Hinit + ((size_t)(b * NHEADS + hd) * NCHUNK + c) * STATE_SZ
                            + p * N_STATE + nc * 8;
    float4 c0 = *(const float4*)(Crow + nc * 8);
    float4 c1 = *(const float4*)(Crow + nc * 8 + 4);
    float4 h0 = *(const float4*)(Hp);
    float4 h1 = *(const float4*)(Hp + 4);
    float acc = c0.x * h0.x + c0.y * h0.y + c0.z * h0.z + c0.w * h0.w
              + c1.x * h1.x + c1.y * h1.y + c1.z * h1.z + c1.w * h1.w;
    acc += __shfl_xor(acc, 1, 32);
    acc += __shfl_xor(acc, 2, 32);
    acc += __shfl_xor(acc, 4, 32);
    if (nc == 0)
        y[(size_t)tok * D_INNER + hd * HEADDIM + p] +=
            cumdec[(size_t)tok * NHEADS + hd] * acc;
}

// v = y_ssm * silu(z); rmsnorm(v, norm_W) -> bf16 (feeds out_proj WMMA). 1 block / token.
__global__ void __launch_bounds__(256)
k_gate_norm(const float* __restrict__ zx, const float* __restrict__ yssm,
            const float* __restrict__ nw, unsigned short* __restrict__ yg) {
    __shared__ float vbuf[D_INNER];
    __shared__ float red[256];
    int t = blockIdx.x;
    const float* zrow = zx + (size_t)t * PROJ_IN;      // z = first D_INNER of zxbcdt
    const float* yrow = yssm + (size_t)t * D_INNER;
    float ss = 0.f;
    for (int d = threadIdx.x; d < D_INNER; d += blockDim.x) {
        float v = yrow[d] * silu_f(zrow[d]);
        vbuf[d] = v;
        ss += v * v;
    }
    ss = block_reduce_sum(ss, red);
    float scale = rsqrtf(ss / (float)D_INNER + 1e-5f);
    unsigned short* orow = yg + (size_t)t * D_INNER;
    for (int d = threadIdx.x; d < D_INNER; d += blockDim.x)
        orow[d] = to_bf16_bits(vbuf[d] * scale * nw[d]);
}

// pooled[b,d] = mean_s xn[b,s,d]
__global__ void k_pool(const float* __restrict__ xn, float* __restrict__ pooled) {
    int idx = blockIdx.x * blockDim.x + threadIdx.x;
    if (idx >= B_SZ * D_MODEL) return;
    int b = idx / D_MODEL, d = idx - b * D_MODEL;
    float s = 0.f;
    for (int t = 0; t < SEQ; ++t)
        s += xn[(size_t)(b * SEQ + t) * D_MODEL + d];
    pooled[idx] = s / (float)SEQ;
}

// out[b,e] = pooled[b,:] @ proj_W[:,e] + proj_b[e]
__global__ void k_proj(const float* __restrict__ pooled, const float* __restrict__ W,
                       const float* __restrict__ bias, float* __restrict__ out) {
    int idx = blockIdx.x * blockDim.x + threadIdx.x;
    if (idx >= B_SZ * EMBED) return;
    int b = idx / EMBED, e = idx - b * EMBED;
    float acc = bias[e];
    const float* pr = pooled + (size_t)b * D_MODEL;
    for (int k = 0; k < D_MODEL; ++k)
        acc += pr[k] * W[(size_t)k * EMBED + e];
    out[idx] = acc;
}

// in-place L2 normalize each row of out[B_SZ x EMBED]
__global__ void __launch_bounds__(256)
k_l2norm(float* __restrict__ out) {
    __shared__ float red[256];
    int b = blockIdx.x;
    float* row = out + (size_t)b * EMBED;
    float vals[EMBED / 256];
    float ss = 0.f;
#pragma unroll
    for (int i = 0; i < EMBED / 256; ++i) {
        float v = row[threadIdx.x + i * 256];
        vals[i] = v;
        ss += v * v;
    }
    ss = block_reduce_sum(ss, red);
    float inv = 1.0f / fmaxf(sqrtf(ss), 1e-12f);
#pragma unroll
    for (int i = 0; i < EMBED / 256; ++i)
        row[threadIdx.x + i * 256] = vals[i] * inv;
}

// ---------------- host launcher ----------------
static inline size_t align256(size_t x) { return (x + 255) & ~(size_t)255; }

extern "C" void kernel_launch(void* const* d_in, const int* in_sizes, int n_in,
                              void* d_out, int out_size, void* d_ws, size_t ws_size,
                              hipStream_t stream) {
    (void)in_sizes; (void)n_in; (void)out_size; (void)ws_size;

    const int*   token_ids = (const int*)  d_in[0];
    const float* embed_W   = (const float*)d_in[1];
    const float* ln_W      = (const float*)d_in[2];
    const float* in_proj_W = (const float*)d_in[3];
    const float* conv_W    = (const float*)d_in[4];
    const float* conv_b    = (const float*)d_in[5];
    const float* dt_bias   = (const float*)d_in[6];
    const float* A_log     = (const float*)d_in[7];
    const float* Dvec      = (const float*)d_in[8];
    const float* norm_W    = (const float*)d_in[9];
    const float* out_proj_W= (const float*)d_in[10];
    const float* final_nW  = (const float*)d_in[11];
    const float* proj_W    = (const float*)d_in[12];
    const float* proj_b    = (const float*)d_in[13];
    float* out = (float*)d_out;

    // workspace carve-out
    char* ws = (char*)d_ws;
    size_t off = 0;
    auto carve = [&](size_t bytes) { char* p = ws + off; off += align256(bytes); return p; };
    float*          x    = (float*)         carve((size_t)NTOK * D_MODEL * 4);
    unsigned short* hbf  = (unsigned short*)carve((size_t)NTOK * D_MODEL * 2);
    unsigned short* w1t  = (unsigned short*)carve((size_t)PROJ_IN * D_MODEL * 2);
    float*          zx   = (float*)         carve((size_t)NTOK * PROJ_IN * 4);
    float*          xbc  = (float*)         carve((size_t)NTOK * CONV_DIM * 4);
    float*          dtb  = (float*)         carve((size_t)NTOK * NHEADS * 4);
    float*          yssm = (float*)         carve((size_t)NTOK * D_INNER * 4);
    unsigned short* yg   = (unsigned short*)carve((size_t)NTOK * D_INNER * 2);
    unsigned short* w2t  = (unsigned short*)carve((size_t)D_MODEL * D_INNER * 2);
    float*          xn   = (float*)         carve((size_t)NTOK * D_MODEL * 4);
    float*          pooled=(float*)         carve((size_t)B_SZ * D_MODEL * 4);
    float*          cumd = (float*)         carve((size_t)NTOK * NHEADS * 4);
    float*          sbuf = (float*)         carve((size_t)B_SZ * NHEADS * NCHUNK * STATE_SZ * 4);

    const int T256 = 256;
    k_embed<<<(NTOK * D_MODEL + T256 - 1) / T256, T256, 0, stream>>>(token_ids, embed_W, x);

    for (int l = 0; l < N_LAYERS; ++l) {
        k_rmsnorm_bf16<<<NTOK, T256, 0, stream>>>(x, ln_W + (size_t)l * D_MODEL, hbf, D_MODEL);
        k_transpose_bf16<<<dim3((PROJ_IN + 31) / 32, (D_MODEL + 31) / 32), T256, 0, stream>>>(
            in_proj_W + (size_t)l * D_MODEL * PROJ_IN, w1t, D_MODEL, PROJ_IN);
        // in_proj GEMM: [2048 x 768] @ [768 x 3248]
        k_gemm_bf16<<<dim3((PROJ_IN + 255) / 256, NTOK / 16), 128, 0, stream>>>(
            hbf, w1t, nullptr, zx, NTOK, PROJ_IN, D_MODEL);
        k_conv_silu<<<(NTOK * CONV_DIM + T256 - 1) / T256, T256, 0, stream>>>(
            zx, conv_W + (size_t)l * CONV_DIM * K_CONV, conv_b + (size_t)l * CONV_DIM, xbc);
        k_dt<<<(NTOK * NHEADS + T256 - 1) / T256, T256, 0, stream>>>(
            zx, dt_bias + (size_t)l * NHEADS, dtb);
        // chunked SSM scan: parallel local scan, tiny combine, parallel fix-up
        k_ssm_chunk<<<B_SZ * NHEADS * NCHUNK, T256, 0, stream>>>(
            xbc, dtb, A_log + (size_t)l * NHEADS, Dvec + (size_t)l * NHEADS, yssm, cumd, sbuf);
        k_ssm_combine<<<B_SZ * NHEADS, T256, 0, stream>>>(sbuf, cumd);
        k_ssm_fix<<<NTOK * NHEADS, T256, 0, stream>>>(xbc, cumd, sbuf, yssm);
        k_gate_norm<<<NTOK, T256, 0, stream>>>(zx, yssm, norm_W + (size_t)l * D_INNER, yg);
        k_transpose_bf16<<<dim3((D_MODEL + 31) / 32, (D_INNER + 31) / 32), T256, 0, stream>>>(
            out_proj_W + (size_t)l * D_INNER * D_MODEL, w2t, D_INNER, D_MODEL);
        // out_proj GEMM with residual: x += yg @ W  ([2048 x 1536] @ [1536 x 768])
        k_gemm_bf16<<<dim3((D_MODEL + 255) / 256, NTOK / 16), 128, 0, stream>>>(
            yg, w2t, x, x, NTOK, D_MODEL, D_INNER);
    }

    k_rmsnorm_f32<<<NTOK, T256, 0, stream>>>(x, final_nW, xn, D_MODEL);
    k_pool<<<(B_SZ * D_MODEL + T256 - 1) / T256, T256, 0, stream>>>(xn, pooled);
    k_proj<<<(B_SZ * EMBED + T256 - 1) / T256, T256, 0, stream>>>(pooled, proj_W, proj_b, out);
    k_l2norm<<<B_SZ, T256, 0, stream>>>(out);
}
